// AntimatterTransformer_81346680586476
// MI455X (gfx1250) — matrix operand order repaired and checked by reading, not compiled, not run on hardware
//
#include <hip/hip_runtime.h>
#include <hip/hip_bf16.h>
#include <math.h>

// ---------------------------------------------------------------------------
// GPT-2-like forward (L=4, B=2, T=1024, C=1024, H=16, V=50257) for gfx1250.
// fp32 residual stream; bf16 WMMA (v_wmma_f32_16x16x32_bf16) for all GEMMs and
// flash attention; GEMM tiles streamed with async global->LDS copies
// (global_load_async_to_lds_b128 / s_wait_asynccnt) and double buffering.
// ---------------------------------------------------------------------------

typedef __attribute__((ext_vector_type(16))) __bf16 v16bf;
typedef __attribute__((ext_vector_type(8)))  float  v8f;

#define LL 4
#define BB 2
#define TT 1024
#define CC 1024
#define HH 16
#define VV 50257
#define DH 64
#define MROWS (BB * TT)     // 2048

// GEMM tiling: block 256 threads = 8 waves (4 m-rows x 2 n-cols),
// tile 64(M) x 128(N) x 32(K); wave owns 16x64 of C -> 4 WMMA per K-step.
#define BM 64
#define BN 128
#define BKT 32
#define ASTR 36             // fp32 LDS stride for A tile rows (32 + 4 pad)
#define BSTR 132            // fp32 LDS stride for B tile rows (128 + 4 pad)
#define ABUF (64 * ASTR)    // 2304 floats
#define BBUF (32 * BSTR)    // 4224 floats

__device__ __forceinline__ __bf16 f2bf(float f) { return (__bf16)f; }

// Async copy 16B global -> LDS (ASYNCcnt path, ISA ch.8).
__device__ __forceinline__ void async_cp_b128(uint32_t lds_off, const float* g) {
  const unsigned long long ga = (unsigned long long)(uintptr_t)g;
  asm volatile("global_load_async_to_lds_b128 %0, %1, off"
               :: "v"(lds_off), "v"(ga) : "memory");
}
__device__ __forceinline__ void wait_async0() {
  asm volatile("s_wait_asynccnt 0" ::: "memory");
}

// A fragment (16x32) from row-major fp32 LDS in this code's contiguous k order:
// lane = M, VGPR r: K = kh + 2r, kh + 2r + 1; kh = (lane>=16)*16.
__device__ __forceinline__ v16bf frag_a_f32(const float* As, int lda) {
  const int lane = threadIdx.x & 31;
  const int m  = lane & 15;
  const int kh = (lane >> 4) << 4;
  v16bf a;
#pragma unroll
  for (int r = 0; r < 8; ++r) {
    const int k = kh + (r << 1);
    a[2 * r]     = f2bf(As[m * lda + k]);
    a[2 * r + 1] = f2bf(As[m * lda + k + 1]);
  }
  return a;
}

// B fragment (32x16) from fp32 LDS stored [K][N]:
// lane = N, koff = (lane>=16)*16, VGPR r: K = koff + 2r, koff + 2r + 1.
__device__ __forceinline__ v16bf frag_b_f32(const float* Bs, int ldb) {
  const int lane = threadIdx.x & 31;
  const int n    = lane & 15;
  const int koff = (lane >> 4) << 4;
  v16bf b;
#pragma unroll
  for (int r = 0; r < 8; ++r) {
    const int k = koff + 2 * r;
    b[2 * r]     = f2bf(Bs[k * ldb + n]);
    b[2 * r + 1] = f2bf(Bs[(k + 1) * ldb + n]);
  }
  return b;
}

// bf16-LDS variants (used by attention's P relayout).
__device__ __forceinline__ v16bf frag_a_lds(const __bf16* As, int lda) {
  const int lane = threadIdx.x & 31;
  const int m  = lane & 15;
  const int kh = (lane >> 4) << 4;
  v16bf a;
#pragma unroll
  for (int r = 0; r < 8; ++r) {
    const int k = kh + (r << 1);
    a[2 * r]     = As[m * lda + k];
    a[2 * r + 1] = As[m * lda + k + 1];
  }
  return a;
}

// ---------------------------------------------------------------------------
// Embedding: x[b,t,c] = wte[idx[b,t],c] + wpe[t,c]
// ---------------------------------------------------------------------------
__global__ __launch_bounds__(256)
void embed_kernel(const int* __restrict__ idx, const float* __restrict__ wte,
                  const float* __restrict__ wpe, float* __restrict__ x) {
  const size_t i = (size_t)blockIdx.x * blockDim.x + threadIdx.x;
  const int c  = (int)(i % CC);
  const int bt = (int)(i / CC);
  const int t  = bt % TT;
  x[i] = wte[(size_t)idx[bt] * CC + c] + wpe[(size_t)t * CC + c];
}

// ---------------------------------------------------------------------------
// LayerNorm: one 256-thread block per row of C=1024.
// ---------------------------------------------------------------------------
__global__ __launch_bounds__(256)
void ln_kernel(const float* __restrict__ x, const float* __restrict__ w,
               const float* __restrict__ b, float* __restrict__ out) {
  __shared__ float rs[256];
  __shared__ float rs2[256];
  const int row = blockIdx.x;
  const float* xr = x + (size_t)row * CC;
  float s = 0.f, s2 = 0.f;
#pragma unroll
  for (int j = 0; j < CC / 256; ++j) {
    const float v = xr[threadIdx.x + j * 256];
    s += v; s2 += v * v;
  }
  rs[threadIdx.x] = s; rs2[threadIdx.x] = s2;
  __syncthreads();
  for (int off = 128; off >= 1; off >>= 1) {
    if ((int)threadIdx.x < off) {
      rs[threadIdx.x]  += rs[threadIdx.x + off];
      rs2[threadIdx.x] += rs2[threadIdx.x + off];
    }
    __syncthreads();
  }
  const float mean = rs[0] * (1.0f / CC);
  const float var  = rs2[0] * (1.0f / CC) - mean * mean;
  const float rstd = rsqrtf(var + 1e-5f);
#pragma unroll
  for (int j = 0; j < CC / 256; ++j) {
    const int i = threadIdx.x + j * 256;
    out[(size_t)row * CC + i] = (xr[i] - mean) * rstd * w[i] + b[i];
  }
}

// ---------------------------------------------------------------------------
// GEMM: out[M,N] = epilogue(A[M,K] x W[K,N] + bias [+ res]).
// Double-buffered async global->LDS staging of raw fp32 tiles; fp32->bf16
// conversion folded into fragment build; 4 WMMA per wave per K-step.
// ACT: 0=none, 1=LeakyReLU(0.01). RES: add residual (same shape as out).
// ---------------------------------------------------------------------------
template <int ACT, bool RES>
__global__ __launch_bounds__(256)
void gemm_kernel(const float* __restrict__ A, const float* __restrict__ W,
                 const float* __restrict__ bias, const float* __restrict__ res,
                 float* __restrict__ out, int M, int N, int K) {
  __shared__ __align__(16) float smem[2 * (ABUF + BBUF)];

  const int tid = threadIdx.x;
  const int m0 = blockIdx.y * BM;
  const int n0 = blockIdx.x * BN;
  const int w    = tid >> 5;
  const int moff = (w & 3) * 16;   // wave m-row
  const int noff = (w >> 2) * 64;  // wave n-col (two waves cover 128)

  const uint32_t smem_base = (uint32_t)(uintptr_t)(void*)smem;

  v8f acc[4] = {};

  // async-issue one 64x32 A tile + 32x128 B tile into buffer p
  auto issue_tile = [&](int p, int kt) {
    const uint32_t base = smem_base + (uint32_t)(p * (ABUF + BBUF) * 4);
#pragma unroll
    for (int i = 0; i < 2; ++i) {            // A: 512 x 16B chunks
      const int c = tid + i * 256;
      const int row = c >> 3, q = (c & 7) * 4;
      async_cp_b128(base + (uint32_t)((row * ASTR + q) * 4),
                    &A[(size_t)(m0 + row) * K + kt + q]);
    }
#pragma unroll
    for (int i = 0; i < 4; ++i) {            // B: 1024 x 16B chunks
      const int c = tid + i * 256;
      const int row = c >> 5, q = (c & 31) * 4;
      async_cp_b128(base + (uint32_t)(ABUF * 4) + (uint32_t)((row * BSTR + q) * 4),
                    &W[(size_t)(kt + row) * N + n0 + q]);
    }
  };

  int p = 0;
  issue_tile(0, 0);

  for (int kt = 0; kt < K; kt += BKT) {
    wait_async0();        // this wave's copies of tile kt complete
    __syncthreads();      // all waves' copies complete
    if (kt + BKT < K) issue_tile(1 - p, kt + BKT);

    const float* Ap = smem + p * (ABUF + BBUF);
    const float* Bp = Ap + ABUF;
    const v16bf a = frag_a_f32(Ap + moff * ASTR, ASTR);
#pragma unroll
    for (int t = 0; t < 4; ++t) {
      const v16bf b = frag_b_f32(Bp + noff + 16 * t, BSTR);
      acc[t] = __builtin_amdgcn_wmma_f32_16x16x32_bf16(false, a, false, b,
                                                       (short)0, acc[t],
                                                       false, false);
    }
    __syncthreads();      // everyone done reading buffer p before re-fill
    p ^= 1;
  }

  // epilogue: C layout = (M = v + 8*(lane>>4), N = lane&15)
  const int lane = tid & 31;
  const int hi = lane >> 4;
  const int nc = lane & 15;
#pragma unroll
  for (int t = 0; t < 4; ++t) {
#pragma unroll
    for (int v = 0; v < 8; ++v) {
      const int row = m0 + moff + v + (hi << 3);
      const int col = n0 + noff + t * 16 + nc;
      float x0 = acc[t][v] + bias[col];
      if (ACT == 1) x0 = (x0 >= 0.f) ? x0 : 0.01f * x0;
      if (RES) x0 += res[(size_t)row * N + col];
      out[(size_t)row * N + col] = x0;
    }
  }
}

// ---------------------------------------------------------------------------
// Flash-style causal attention. One wave per (b, h, 16-query tile).
// S = Q K^T (2x2 WMMA over a 32-key block), fp32 online softmax with
// __shfl_xor row reductions, P relaid via LDS into A-fragment form,
// O += P V (4 WMMA). qkv layout [B,T,3C]; y layout [B,T,C].
// ---------------------------------------------------------------------------
__global__ __launch_bounds__(32)
void attn_kernel(const float* __restrict__ qkv, float* __restrict__ y) {
  __shared__ __bf16 Ps[16 * 34];

  const int lane = threadIdx.x & 31;
  const int b = blockIdx.z;
  const int h = blockIdx.y;
  const int qbase = blockIdx.x * 16;
  const float scale = 0.125f;  // 1/sqrt(64)
  const size_t rstr = 3 * CC;

  const float* qp = qkv + (size_t)b * TT * rstr + h * DH;
  const float* kp = qp + CC;
  const float* vp = qp + 2 * CC;

  const int mA = lane & 15;
  const int hi = lane >> 4;
  const int khA = hi << 4;
  const int nB = lane & 15;
  const int koffB = hi << 4;

  // Q fragments: A[m][k] = Q[qbase+m][kk*32 + k]
  v16bf aq[2];
#pragma unroll
  for (int kk = 0; kk < 2; ++kk) {
#pragma unroll
    for (int r = 0; r < 8; ++r) {
      const int k = khA + (r << 1);
      const float* src = qp + (size_t)(qbase + mA) * rstr + kk * 32 + k;
      aq[kk][2 * r]     = f2bf(src[0]);
      aq[kk][2 * r + 1] = f2bf(src[1]);
    }
  }

  float m_i[8], l_i[8];
  v8f o[4] = {};
#pragma unroll
  for (int v = 0; v < 8; ++v) { m_i[v] = -1e30f; l_i[v] = 0.f; }

  const int jmax = (qbase + 15) >> 5;
  for (int j = 0; j <= jmax; ++j) {
    const int kb = j << 5;

    // S = Q K^T : B[k][n] = K[kb + nsub*16 + n][kk*32 + k]
    v8f s0 = {};
    v8f s1 = {};
#pragma unroll
    for (int kk = 0; kk < 2; ++kk) {
      v16bf bk0, bk1;
#pragma unroll
      for (int r = 0; r < 8; ++r) {
        const int kd = koffB + 2 * r;
        const float* p0 = kp + (size_t)(kb + nB) * rstr + kk * 32 + kd;
        const float* p1 = kp + (size_t)(kb + 16 + nB) * rstr + kk * 32 + kd;
        bk0[2 * r]     = f2bf(p0[0]);
        bk0[2 * r + 1] = f2bf(p0[1]);
        bk1[2 * r]     = f2bf(p1[0]);
        bk1[2 * r + 1] = f2bf(p1[1]);
      }
      s0 = __builtin_amdgcn_wmma_f32_16x16x32_bf16(false, aq[kk], false, bk0,
                                                   (short)0, s0, false, false);
      s1 = __builtin_amdgcn_wmma_f32_16x16x32_bf16(false, aq[kk], false, bk1,
                                                   (short)0, s1, false, false);
    }

    // scale + causal mask (C layout: row = v + 8*hi, col = lane&15)
    float sv0[8], sv1[8];
    const int key0 = kb + (lane & 15);
    const int key1 = key0 + 16;
#pragma unroll
    for (int v = 0; v < 8; ++v) {
      const int qrow = qbase + v + (hi << 3);
      sv0[v] = (key0 <= qrow) ? s0[v] * scale : -1e30f;
      sv1[v] = (key1 <= qrow) ? s1[v] * scale : -1e30f;
    }

    // online softmax (xor masks <16 keep the two row halves separate)
#pragma unroll
    for (int v = 0; v < 8; ++v) {
      float t = fmaxf(sv0[v], sv1[v]);
#pragma unroll
      for (int off = 8; off >= 1; off >>= 1)
        t = fmaxf(t, __shfl_xor(t, off, 32));
      const float mn = fmaxf(m_i[v], t);
      const float alpha = __expf(m_i[v] - mn);
      m_i[v] = mn;
      sv0[v] = __expf(sv0[v] - mn);
      sv1[v] = __expf(sv1[v] - mn);
      float ls = sv0[v] + sv1[v];
#pragma unroll
      for (int off = 8; off >= 1; off >>= 1)
        ls += __shfl_xor(ls, off, 32);
      l_i[v] = l_i[v] * alpha + ls;
#pragma unroll
      for (int dc = 0; dc < 4; ++dc) o[dc][v] *= alpha;
    }

    // P (C layout) -> LDS -> A-fragment layout
#pragma unroll
    for (int v = 0; v < 8; ++v) {
      const int pr = v + (hi << 3);
      Ps[pr * 34 + (lane & 15)]      = f2bf(sv0[v]);
      Ps[pr * 34 + 16 + (lane & 15)] = f2bf(sv1[v]);
    }
    __syncthreads();
    const v16bf ap = frag_a_lds(Ps, 34);
    __syncthreads();

    // O += P V : B[k][n] = V[kb + k][dc*16 + n]
#pragma unroll
    for (int dc = 0; dc < 4; ++dc) {
      v16bf bv;
#pragma unroll
      for (int r = 0; r < 8; ++r) {
        const int kd = koffB + 2 * r;
        bv[2 * r]     = f2bf(vp[(size_t)(kb + kd) * rstr + dc * 16 + nB]);
        bv[2 * r + 1] = f2bf(vp[(size_t)(kb + kd + 1) * rstr + dc * 16 + nB]);
      }
      o[dc] = __builtin_amdgcn_wmma_f32_16x16x32_bf16(false, ap, false, bv,
                                                      (short)0, o[dc], false, false);
    }
  }

  // normalize + store y[b,t, h*64 + d]
#pragma unroll
  for (int v = 0; v < 8; ++v) {
    const int t = qbase + v + (hi << 3);
    const float rinv = 1.0f / l_i[v];
    float* dst = y + ((size_t)(b * TT + t)) * CC + h * DH;
#pragma unroll
    for (int dc = 0; dc < 4; ++dc)
      dst[dc * 16 + (lane & 15)] = o[dc][v] * rinv;
  }
}

// ---------------------------------------------------------------------------
// lm_head: logits[b, v] = dot(h_final[b, T-1, :], wte[v, :]). One wave per v.
// ---------------------------------------------------------------------------
__global__ __launch_bounds__(256)
void lmhead_kernel(const float* __restrict__ hfin, const float* __restrict__ wte,
                   float* __restrict__ out) {
  const int lane = threadIdx.x & 31;
  const int v = blockIdx.x * 8 + (threadIdx.x >> 5);
  if (v >= VV) return;
  const float* wrow = wte + (size_t)v * CC;
  const float* h0 = hfin + (size_t)(0 * TT + TT - 1) * CC;
  const float* h1 = hfin + (size_t)(1 * TT + TT - 1) * CC;
  float a0 = 0.f, a1 = 0.f;
#pragma unroll
  for (int i = 0; i < CC / 32; ++i) {
    const float wv = wrow[lane + i * 32];
    a0 += h0[lane + i * 32] * wv;
    a1 += h1[lane + i * 32] * wv;
  }
  for (int off = 16; off >= 1; off >>= 1) {
    a0 += __shfl_xor(a0, off, 32);
    a1 += __shfl_xor(a1, off, 32);
  }
  if (lane == 0) { out[v] = a0; out[VV + v] = a1; }
}

// ---------------------------------------------------------------------------
extern "C" void kernel_launch(void* const* d_in, const int* in_sizes, int n_in,
                              void* d_out, int out_size, void* d_ws, size_t ws_size,
                              hipStream_t stream) {
  const int*   idx    = (const int*)  d_in[0];
  const float* wte    = (const float*)d_in[1];
  const float* wpe    = (const float*)d_in[2];
  const float* ln1_w  = (const float*)d_in[3];
  const float* ln1_b  = (const float*)d_in[4];
  const float* attn_w = (const float*)d_in[5];
  const float* attn_b = (const float*)d_in[6];
  const float* proj_w = (const float*)d_in[7];
  const float* proj_b = (const float*)d_in[8];
  const float* ln2_w  = (const float*)d_in[9];
  const float* ln2_b  = (const float*)d_in[10];
  const float* fc_w   = (const float*)d_in[11];
  const float* fc_b   = (const float*)d_in[12];
  const float* fcp_w  = (const float*)d_in[13];
  const float* fcp_b  = (const float*)d_in[14];
  const float* lnf_w  = (const float*)d_in[15];
  const float* lnf_b  = (const float*)d_in[16];
  float* out = (float*)d_out;

  // workspace carve-out (80 MB total)
  float* x    = (float*)d_ws;                       // [2048,1024]
  float* hbuf = x    + (size_t)MROWS * CC;          // [2048,1024]
  float* qkv  = hbuf + (size_t)MROWS * CC;          // [2048,3072]
  float* ybuf = qkv  + (size_t)MROWS * 3 * CC;      // [2048,1024]
  float* fch  = ybuf + (size_t)MROWS * CC;          // [2048,4096]

  embed_kernel<<<(MROWS * CC) / 256, 256, 0, stream>>>(idx, wte, wpe, x);

  for (int l = 0; l < LL; ++l) {
    ln_kernel<<<MROWS, 256, 0, stream>>>(x, ln1_w + l * CC, ln1_b + l * CC, hbuf);
    gemm_kernel<0, false><<<dim3(3 * CC / BN, MROWS / BM), 256, 0, stream>>>(
        hbuf, attn_w + (size_t)l * CC * 3 * CC, attn_b + (size_t)l * 3 * CC,
        nullptr, qkv, MROWS, 3 * CC, CC);
    attn_kernel<<<dim3(TT / 16, HH, BB), 32, 0, stream>>>(qkv, ybuf);
    gemm_kernel<0, true><<<dim3(CC / BN, MROWS / BM), 256, 0, stream>>>(
        ybuf, proj_w + (size_t)l * CC * CC, proj_b + (size_t)l * CC,
        x, x, MROWS, CC, CC);
    ln_kernel<<<MROWS, 256, 0, stream>>>(x, ln2_w + l * CC, ln2_b + l * CC, hbuf);
    gemm_kernel<1, false><<<dim3(4 * CC / BN, MROWS / BM), 256, 0, stream>>>(
        hbuf, fc_w + (size_t)l * CC * 4 * CC, fc_b + (size_t)l * 4 * CC,
        nullptr, fch, MROWS, 4 * CC, CC);
    gemm_kernel<0, true><<<dim3(CC / BN, MROWS / BM), 256, 0, stream>>>(
        fch, fcp_w + (size_t)l * 4 * CC * CC, fcp_b + (size_t)l * CC,
        x, x, MROWS, CC, 4 * CC);
  }

  ln_kernel<<<MROWS, 256, 0, stream>>>(x, lnf_w, lnf_b, hbuf);
  lmhead_kernel<<<(VV + 7) / 8, 256, 0, stream>>>(hbuf, wte, out);
}
